// HybridLSTMTagger_65481071402005
// MI455X (gfx1250) — matrix operand order repaired and checked by reading, not compiled, not run on hardware
//
#include <hip/hip_runtime.h>
#include <hip/hip_bf16.h>

typedef __attribute__((ext_vector_type(16))) _Float16 v16h;
typedef __attribute__((ext_vector_type(8)))  _Float16 v8h;
typedef __attribute__((ext_vector_type(8)))  float    v8f;

#define SEQ   512
#define BATCH 512
#define EMB   256
#define NQ    8
#define TAGS  50
#define DIN   264   // EMB + HID(8)

// ---------------- dependency masks (constexpr mirror of _dep_matrix(8,2)) ---
struct DepMasks { unsigned m[NQ]; };
constexpr DepMasks make_dep() {
    DepMasks d{};
    bool M[NQ][NQ] = {};
    for (int i = 0; i < NQ; ++i) M[i][i] = true;
    for (int dpt = 0; dpt < 2; ++dpt) {
        for (int i = 0; i < NQ - 1; ++i)
            for (int j = 0; j < NQ; ++j) M[i + 1][j] ^= M[i][j];
        for (int j = 0; j < NQ; ++j) M[0][j] ^= M[NQ - 1][j];
    }
    for (int i = 0; i < NQ; ++i) {
        unsigned mm = 0;
        for (int j = 0; j < NQ; ++j) if (M[i][j]) mm |= (1u << j);
        d.m[i] = mm;
    }
    return d;
}
constexpr DepMasks DEPC = make_dep();
__constant__ unsigned DEPM[NQ] = { DEPC.m[0], DEPC.m[1], DEPC.m[2], DEPC.m[3],
                                   DEPC.m[4], DEPC.m[5], DEPC.m[6], DEPC.m[7] };

__device__ __forceinline__ float fsigmoid(float x) { return 1.0f / (1.0f + __expf(-x)); }
__device__ __forceinline__ float ftanh(float x)    { float e = __expf(2.0f * x); return (e - 1.0f) / (e + 1.0f); }

// ---------------- 1. gather embeddings (col 0) -> f16, pack weights -> f16 --
__global__ __launch_bounds__(256) void prep_kernel(
    const int* __restrict__ sentence, const float* __restrict__ emb,
    const float* __restrict__ Wf, const float* __restrict__ Wi,
    const float* __restrict__ Wg, const float* __restrict__ Wo,
    _Float16* __restrict__ Xh, _Float16* __restrict__ Wh)
{
    int b = blockIdx.x;
    int tid = threadIdx.x;
    if (b < SEQ) {
        long row = (long)sentence[b * BATCH];           // sentence[t, 0]
        Xh[b * EMB + tid] = (_Float16)emb[row * EMB + tid];
    } else {
        int gate = b - SEQ;                             // 0..3
        const float* W = (gate == 0) ? Wf : (gate == 1) ? Wi : (gate == 2) ? Wg : Wo;
        for (int q = 0; q < NQ; ++q) {
            int g = gate * NQ + q;                      // 0..31
            Wh[g * EMB + tid] = (_Float16)W[q * DIN + tid];   // x-part cols 0..255
        }
    }
}

// ---------------- 2. WMMA GEMM: pre[t][g] = sum_k Xh[t][k]*Wh[g][k] ---------
// M=512 (t), N=32 (g = gate*8+q), K=256.  One block per 16-row M-tile,
// two waves per block: wave w owns N-tile w (g 0-15 / 16-31).
__global__ __launch_bounds__(64) void gemm_kernel(
    const _Float16* __restrict__ Xh, const _Float16* __restrict__ Wh,
    float* __restrict__ pre)
{
    int mtile = blockIdx.x;            // 0..31
    int wave  = threadIdx.x >> 5;      // 0..1  (N-tile)
    int lane  = threadIdx.x & 31;
    int lm    = lane & 15;
    int hi    = lane >> 4;             // lane half
    int t     = mtile * 16 + lm;       // A row for this lane (both halves: M = lane%16)
    int g     = wave * 16 + lm;        // B column for this lane

    v8f acc = {};
    #pragma unroll
    for (int kb = 0; kb < EMB; kb += 32) {
        // A 16x32 f16 frag (ISA 7.12.2): lanes<16 halves = K{0..7,16..23},
        // lanes>=16 halves = K{8..15,24..31}
        union { v16h v; v8h h[2]; } a;
        a.h[0] = *(const v8h*)(Xh + t * EMB + kb + hi * 8);
        a.h[1] = *(const v8h*)(Xh + t * EMB + kb + 16 + hi * 8);
        // B 32x16 f16 frag: lane = N, lanes<16 halves = K{0..7,16..23}, lanes>=16 = K{8..15,24..31}
        union { v16h v; v8h h[2]; } b;
        b.h[0] = *(const v8h*)(Wh + g * EMB + kb + hi * 8);
        b.h[1] = *(const v8h*)(Wh + g * EMB + kb + 16 + hi * 8);
        acc = __builtin_amdgcn_wmma_f32_16x16x32_f16(
            /*neg_a=*/false, a.v, /*neg_b=*/false, b.v,
            /*c_mod=*/(short)0, acc, /*reuse_a=*/false, /*reuse_b=*/false);
    }
    // C/D layout: lane gives N (=lm), VGPR r gives M = r + hi*8
    #pragma unroll
    for (int r = 0; r < 8; ++r) {
        int tt = mtile * 16 + r + hi * 8;
        pre[tt * 32 + g] = acc[r];
    }
}

// ---------------- 3. sequential LSTM scan, one wave32 ----------------------
// lane = gate*8 + q.  Recurrent part is only the 8 h-columns of each W.
__global__ __launch_bounds__(32) void scan_kernel(
    const float* __restrict__ pre,
    const float* __restrict__ Wf, const float* __restrict__ bf,
    const float* __restrict__ Wi, const float* __restrict__ bi,
    const float* __restrict__ Wg, const float* __restrict__ bg,
    const float* __restrict__ Wo, const float* __restrict__ bo,
    float* __restrict__ hs)
{
    int lane = threadIdx.x;
    int gate = lane >> 3;
    int q    = lane & 7;
    const float* W = (gate == 0) ? Wf : (gate == 1) ? Wi : (gate == 2) ? Wg : Wo;
    const float* B = (gate == 0) ? bf : (gate == 1) ? bi : (gate == 2) ? bg : bo;
    float wh[NQ];
    #pragma unroll
    for (int j = 0; j < NQ; ++j) wh[j] = W[q * DIN + EMB + j];   // h-part cols 256..263
    float bias = B[q];
    unsigned mask = DEPM[q];

    float hx = 0.0f, cx = 0.0f;      // valid per q in every gate-group (kept identical)
    for (int t = 0; t < SEQ; ++t) {
        float z = pre[t * 32 + lane] + bias;
        #pragma unroll
        for (int j = 0; j < NQ; ++j) z += wh[j] * __shfl(hx, j, 32);
        float c = __cosf(z);
        int base = lane & ~7;
        float p = 1.0f;
        #pragma unroll
        for (int j = 0; j < NQ; ++j) {
            float cj = __shfl(c, base + j, 32);
            if ((mask >> j) & 1u) p *= cj;
        }
        float act = (gate == 2) ? ftanh(p) : fsigmoid(p);
        // gather f,i,g,o for this qubit from the four gate groups
        float fv = __shfl(act,      q, 32);
        float iv = __shfl(act,  8 + q, 32);
        float gv = __shfl(act, 16 + q, 32);
        float ov = __shfl(act, 24 + q, 32);
        cx = fv * cx + iv * gv;
        hx = ov * ftanh(cx);
        if (lane < NQ) hs[t * NQ + lane] = hx;
    }
}

// ---------------- 4. tagger + log_softmax ----------------------------------
__global__ __launch_bounds__(128) void tag_kernel(
    const float* __restrict__ hs, const float* __restrict__ Wtag,
    const float* __restrict__ btag, float* __restrict__ logp)
{
    __shared__ float sW[TAGS * NQ];
    __shared__ float sb[TAGS];
    for (int i = threadIdx.x; i < TAGS * NQ; i += blockDim.x) sW[i] = Wtag[i];
    for (int i = threadIdx.x; i < TAGS;      i += blockDim.x) sb[i] = btag[i];
    __syncthreads();
    int t = blockIdx.x * blockDim.x + threadIdx.x;   // 0..511
    float h[NQ];
    #pragma unroll
    for (int j = 0; j < NQ; ++j) h[j] = hs[t * NQ + j];
    float lg[TAGS];
    float mx = -3.4e38f;
    #pragma unroll
    for (int k = 0; k < TAGS; ++k) {
        float s = sb[k];
        #pragma unroll
        for (int j = 0; j < NQ; ++j) s += sW[k * NQ + j] * h[j];
        lg[k] = s;
        mx = fmaxf(mx, s);
    }
    float sum = 0.0f;
    #pragma unroll
    for (int k = 0; k < TAGS; ++k) sum += __expf(lg[k] - mx);
    float lse = mx + __logf(sum);
    #pragma unroll
    for (int k = 0; k < TAGS; ++k) logp[t * TAGS + k] = lg[k] - lse;
}

// ---------------- 5. broadcast (t,50) -> (t,512,50): HBM-bound -------------
__global__ __launch_bounds__(256) void bcast_kernel(
    const float* __restrict__ logp, float* __restrict__ out)
{
    __shared__ float row[TAGS];
    int t = blockIdx.y;
    if (threadIdx.x < TAGS) row[threadIdx.x] = logp[t * TAGS + threadIdx.x];
    __syncthreads();
    int off = (blockIdx.x * 256 + threadIdx.x) * 4;   // 0..25596 within slice
    int k = off % TAGS;
    float4 v;
    v.x = row[k]; k = (k + 1 == TAGS) ? 0 : k + 1;
    v.y = row[k]; k = (k + 1 == TAGS) ? 0 : k + 1;
    v.z = row[k]; k = (k + 1 == TAGS) ? 0 : k + 1;
    v.w = row[k];
    *(float4*)(out + (size_t)t * (BATCH * TAGS) + off) = v;
}

// ---------------- launch ---------------------------------------------------
extern "C" void kernel_launch(void* const* d_in, const int* in_sizes, int n_in,
                              void* d_out, int out_size, void* d_ws, size_t ws_size,
                              hipStream_t stream) {
    const int*   sentence = (const int*)  d_in[0];
    const float* emb      = (const float*)d_in[1];
    const float* W_f = (const float*)d_in[2];  const float* b_f = (const float*)d_in[3];
    const float* W_i = (const float*)d_in[4];  const float* b_i = (const float*)d_in[5];
    const float* W_g = (const float*)d_in[6];  const float* b_g = (const float*)d_in[7];
    const float* W_o = (const float*)d_in[8];  const float* b_o = (const float*)d_in[9];
    // d_in[10..13] = p_f..p_o : unused by the reference forward pass
    const float* W_tag = (const float*)d_in[14];
    const float* b_tag = (const float*)d_in[15];
    float* out = (float*)d_out;

    char* ws = (char*)d_ws;
    _Float16* Xh  = (_Float16*)(ws);                       // 512*256*2   = 262144 B
    _Float16* Wh  = (_Float16*)(ws + 262144);              // 32*256*2    =  16384 B
    float*    pre = (float*)   (ws + 278528);              // 512*32*4    =  65536 B
    float*    hsb = (float*)   (ws + 344064);              // 512*8*4     =  16384 B
    float*    lgp = (float*)   (ws + 360448);              // 512*50*4    = 102400 B

    prep_kernel<<<SEQ + 4, 256, 0, stream>>>(sentence, emb, W_f, W_i, W_g, W_o, Xh, Wh);
    gemm_kernel<<<SEQ / 16, 64, 0, stream>>>(Xh, Wh, pre);
    scan_kernel<<<1, 32, 0, stream>>>(pre, W_f, b_f, W_i, b_i, W_g, b_g, W_o, b_o, hsb);
    tag_kernel<<<SEQ / 128, 128, 0, stream>>>(hsb, W_tag, b_tag, lgp);
    bcast_kernel<<<dim3((BATCH * TAGS) / (256 * 4), SEQ), 256, 0, stream>>>(lgp, out);
}
